// SinkhornAttentionWithConv_56538949485015
// MI455X (gfx1250) — matrix operand order repaired and checked by reading, not compiled, not run on hardware
//
#include <hip/hip_runtime.h>

#define BATCH 2
#define DCH   256
#define NPIX  4096   // 64*64
#define OCH   768    // 3*DCH

typedef __attribute__((ext_vector_type(16))) _Float16 v16h;
typedef __attribute__((ext_vector_type(8)))  _Float16 v8h;
typedef __attribute__((ext_vector_type(8)))  float    v8f;
typedef __attribute__((ext_vector_type(4)))  float    v4f;
typedef __attribute__((ext_vector_type(4)))  unsigned int u32x4;
typedef __attribute__((ext_vector_type(8)))  int      i32x8;
typedef __attribute__((ext_vector_type(4)))  int      i32x4;

#if __has_builtin(__builtin_amdgcn_tensor_load_to_lds) && \
    __has_builtin(__builtin_amdgcn_s_wait_tensorcnt)
#define HAVE_TDM 1
#else
#define HAVE_TDM 0
#endif

// ---- WMMA fragment layouts used in this file (16-bit 16x16x32; A/B k order is contiguous, not ISA table order) ----
// A (16x32, MxK): row M = lane&15 ; element j -> K = j + (lane>=16?16:0)
// B (32x16, KxN): col N = lane&15 ; element j -> K = j + (lane>=16?16:0)
// C/D (16x16 f32): col = lane&15 ; VGPR r -> row = r + (lane>=16 ? 8 : 0)

__device__ __forceinline__ v16h cat8(v8h lo, v8h hi) {
  return __builtin_shufflevector(lo, hi, 0, 1, 2, 3, 4, 5, 6, 7,
                                 8, 9, 10, 11, 12, 13, 14, 15);
}
__device__ __forceinline__ v16h loadAfrag(const _Float16* row, int kk, int lane) {
  const int base = kk * 32 + ((lane >= 16) ? 16 : 0);
  return cat8(*(const v8h*)(row + base), *(const v8h*)(row + base + 8));
}
__device__ __forceinline__ v16h loadBfrag(const _Float16* row, int kk, int lane) {
  return *(const v16h*)(row + kk * 32 + ((lane >= 16) ? 16 : 0));
}
__device__ __forceinline__ v16h loadAfrag_f32(const float* row, int kk, int lane) {
  const int base = kk * 32 + ((lane >= 16) ? 16 : 0);
  v4f a0 = *(const v4f*)(row + base);
  v4f a1 = *(const v4f*)(row + base + 4);
  v4f a2 = *(const v4f*)(row + base + 8);
  v4f a3 = *(const v4f*)(row + base + 12);
  v16h r;
#pragma unroll
  for (int i = 0; i < 4; ++i) {
    r[i]      = (_Float16)a0[i];
    r[4 + i]  = (_Float16)a1[i];
    r[8 + i]  = (_Float16)a2[i];
    r[12 + i] = (_Float16)a3[i];
  }
  return r;
}
__device__ __forceinline__ v8f wmma_f16(v16h a, v16h b, v8f c) {
  return __builtin_amdgcn_wmma_f32_16x16x32_f16(false, a, false, b, (short)0, c,
                                                false, false);
}

// ---- Tensor Data Mover: DMA a [rows x 256] f16 tile (row-major, stride 256)
// from global into LDS.  D# built per CDNA5 ISA.
__device__ __forceinline__ void tdm_load_tile(const _Float16* gsrc, void* lds_dst,
                                              int rows) {
#if HAVE_TDM
  const unsigned long long ga = (unsigned long long)(size_t)gsrc;   // byte addr
  const unsigned int lds = (unsigned int)(size_t)lds_dst;           // LDS offset
  u32x4 g0;
  g0[0] = 1u;                                        // count=1 (valid), user mode
  g0[1] = lds;                                       // lds_addr [63:32]
  g0[2] = (unsigned int)(ga & 0xFFFFFFFFu);          // global_addr [95:64]
  g0[3] = (unsigned int)((ga >> 32) & 0x01FFFFFFu)   // global_addr [120:96]
          | (2u << 30);                              // type=2 ("image")
  i32x8 g1;
  g1[0] = (int)(1u << 16);          // data_size=1 (2 bytes)
  g1[1] = (int)(DCH << 16);         // tensor_dim0[15:0] in [63:48]
  g1[2] = (int)(NPIX << 16);        // tensor_dim0 hi=0 | tensor_dim1[15:0]
  g1[3] = (int)(DCH << 16);         // tensor_dim1 hi=0 | tile_dim0=256
  g1[4] = rows;                     // tile_dim1=rows | tile_dim2=0
  g1[5] = DCH;                      // tensor_dim0_stride[31:0]=256
  g1[6] = 0;                        // stride0 hi | tensor_dim1_stride lo
  g1[7] = 0;
  i32x4 zz = {0, 0, 0, 0};
#if __has_include(<hip/amd_detail/amd_gfx1250_TDM.h>)
  i32x8 z8 = {0, 0, 0, 0, 0, 0, 0, 0};
  __builtin_amdgcn_tensor_load_to_lds(g0, g1, zz, zz, z8, 0);
#else
  __builtin_amdgcn_tensor_load_to_lds(g0, g1, zz, zz, 0);
#endif
#endif
}
__device__ __forceinline__ void tdm_wait() {
#if HAVE_TDM
  __builtin_amdgcn_s_wait_tensorcnt(0);
#endif
}

// =====================================================================
// Kernel 1: 1x1-conv QKV projection via WMMA.
// q,k written [B][N][D] (pre-scaled by 0.25 = 1/d^0.25); v written
// TRANSPOSED [B][D][N] so pass-3 V fragments are contiguous vector loads.
// =====================================================================
__global__ void sinkq_qkv(const float* __restrict__ x, const float* __restrict__ Wt,
                          const float* __restrict__ bias,
                          _Float16* __restrict__ qh, _Float16* __restrict__ kh,
                          _Float16* __restrict__ vt) {
  const int lane = threadIdx.x & 31;
  const int wave = threadIdx.x >> 5;
  const int o0 = blockIdx.x * 128 + wave * 16;   // output-channel tile
  const int p0 = blockIdx.y * 16;                // global pixel tile
  const int b  = p0 >> 12;
  const int n0 = p0 & (NPIX - 1);
  const int rc = lane & 15;

  v8f acc = {};
#pragma unroll
  for (int kk = 0; kk < 8; ++kk) {
    v16h a = loadAfrag_f32(Wt + (size_t)(o0 + rc) * DCH, kk, lane);
    v16h bf;
#pragma unroll
    for (int j = 0; j < 16; ++j)
      bf[j] = (_Float16)x[(size_t)b * DCH * NPIX +
                          (size_t)(kk * 32 + j + ((lane >= 16) ? 16 : 0)) * NPIX +
                          (n0 + rc)];
    acc = wmma_f16(a, bf, acc);
  }

  const int rowoff = (lane >= 16) ? 8 : 0;
#pragma unroll
  for (int r = 0; r < 8; ++r) {
    const int o = o0 + r + rowoff;
    const int n = n0 + rc;
    const float val = acc[r] + bias[o];
    if (o < DCH)
      qh[(size_t)b * NPIX * DCH + (size_t)n * DCH + o] = (_Float16)(val * 0.25f);
    else if (o < 2 * DCH)
      kh[(size_t)b * NPIX * DCH + (size_t)n * DCH + (o - DCH)] = (_Float16)(val * 0.25f);
    else  // v transposed: [B][D][N]
      vt[(size_t)b * DCH * NPIX + (size_t)(o - 2 * DCH) * NPIX + n] = (_Float16)val;
  }
}

// =====================================================================
// Kernel 2: row LSE of negC = q k^T -> u[b][n]  (flash pass 1).
// Block = 8 waves x 16 rows; shared 16-row K tile staged into LDS,
// double-buffered via TDM (fallback: cooperative vector copy).
// =====================================================================
__global__ void sinkq_row_lse(const _Float16* __restrict__ qh,
                              const _Float16* __restrict__ kh,
                              float* __restrict__ u) {
  __shared__ __align__(32) _Float16 ktile[2][16 * DCH];   // 2 x 8KB

  const int tid  = threadIdx.x;
  const int lane = tid & 31;
  const int wave = tid >> 5;
  const int g  = blockIdx.x * 8 + wave;
  const int p0 = g * 16;
  const int b  = p0 >> 12;            // whole block is in one batch
  const int n0 = p0 & (NPIX - 1);
  const int rc = lane & 15;
  const size_t qb = (size_t)b * NPIX * DCH;

  const _Float16* qrow = qh + qb + (size_t)(n0 + rc) * DCH;
  v16h aq[8];
#pragma unroll
  for (int kk = 0; kk < 8; ++kk) aq[kk] = loadAfrag(qrow, kk, lane);

  float mr[8], rr[8];
#pragma unroll
  for (int r = 0; r < 8; ++r) { mr[r] = -1e30f; rr[r] = 0.f; }

#if HAVE_TDM
  if (wave == 0) tdm_load_tile(kh + qb, &ktile[0][0], 16);
#else
  *(v16h*)&ktile[0][tid * 16] = *(const v16h*)&kh[qb + tid * 16];
#endif

  for (int mt = 0; mt < NPIX / 16; ++mt) {
#if HAVE_TDM
    if (wave == 0) tdm_wait();          // tile (mt&1) landed in LDS
#endif
    __syncthreads();                    // broadcast readiness; prev reads done
    if (mt + 1 < NPIX / 16) {
#if HAVE_TDM
      if (wave == 0)
        tdm_load_tile(kh + qb + (size_t)(mt + 1) * 16 * DCH,
                      &ktile[(mt + 1) & 1][0], 16);
#else
      const _Float16* src = kh + qb + (size_t)(mt + 1) * 16 * DCH;
      *(v16h*)&ktile[(mt + 1) & 1][tid * 16] = *(const v16h*)&src[tid * 16];
      __builtin_prefetch(&kh[qb + (size_t)(mt + 2) * 16 * DCH + tid * 16], 0, 0);
#endif
    }
    const _Float16* krow = &ktile[mt & 1][rc * DCH];
    v8f s = {};
#pragma unroll
    for (int kk = 0; kk < 8; ++kk)
      s = wmma_f16(aq[kk], loadBfrag(krow, kk, lane), s);
#pragma unroll
    for (int r = 0; r < 8; ++r) {          // online LSE, slot r = fixed row
      const float sv = s[r];
      const float mn = fmaxf(mr[r], sv);
      rr[r] = rr[r] * __expf(mr[r] - mn) + __expf(sv - mn);
      mr[r] = mn;
    }
  }

  const float logm = logf(1.0f / (float)NPIX + 1e-8f);
#pragma unroll
  for (int r = 0; r < 8; ++r) {
    float m = mr[r], rs = rr[r];
#pragma unroll
    for (int off = 1; off < 16; off <<= 1) {   // reduce across 16 cols
      const float m2 = __shfl_xor(m, off, 32);
      const float r2 = __shfl_xor(rs, off, 32);
      const float mn = fmaxf(m, m2);
      rs = rs * __expf(m - mn) + r2 * __expf(m2 - mn);
      m = mn;
    }
    if (rc == 0) {
      const int n = n0 + r + ((lane >= 16) ? 8 : 0);
      u[b * NPIX + n] = logm - (m + logf(rs));
    }
  }
}

// =====================================================================
// Kernel 3: column LSE of negC + u[n] -> vpot[b][m]  (flash pass 2).
// Block = 8 waves x 16 columns; shared 16-row Q tile staged via TDM.
// =====================================================================
__global__ void sinkq_col_lse(const _Float16* __restrict__ qh,
                              const _Float16* __restrict__ kh,
                              const float* __restrict__ u,
                              float* __restrict__ vp) {
  __shared__ __align__(32) _Float16 qtile[2][16 * DCH];

  const int tid  = threadIdx.x;
  const int lane = tid & 31;
  const int wave = tid >> 5;
  const int g  = blockIdx.x * 8 + wave;
  const int p0 = g * 16;
  const int b  = p0 >> 12;
  const int m0 = p0 & (NPIX - 1);
  const int rc = lane & 15;
  const int rowoff = (lane >= 16) ? 8 : 0;
  const size_t qb = (size_t)b * NPIX * DCH;

  v16h bk[8];                                  // K fragments fixed per wave
#pragma unroll
  for (int kk = 0; kk < 8; ++kk)
    bk[kk] = loadBfrag(kh + qb + (size_t)(m0 + rc) * DCH, kk, lane);

  float mc = -1e30f, rs = 0.f;

#if HAVE_TDM
  if (wave == 0) tdm_load_tile(qh + qb, &qtile[0][0], 16);
#else
  *(v16h*)&qtile[0][tid * 16] = *(const v16h*)&qh[qb + tid * 16];
#endif

  for (int nt = 0; nt < NPIX / 16; ++nt) {
#if HAVE_TDM
    if (wave == 0) tdm_wait();
#endif
    __syncthreads();
    if (nt + 1 < NPIX / 16) {
#if HAVE_TDM
      if (wave == 0)
        tdm_load_tile(qh + qb + (size_t)(nt + 1) * 16 * DCH,
                      &qtile[(nt + 1) & 1][0], 16);
#else
      const _Float16* src = qh + qb + (size_t)(nt + 1) * 16 * DCH;
      *(v16h*)&qtile[(nt + 1) & 1][tid * 16] = *(const v16h*)&src[tid * 16];
      __builtin_prefetch(&qh[qb + (size_t)(nt + 2) * 16 * DCH + tid * 16], 0, 0);
#endif
    }
    const int n0 = nt * 16;
    const _Float16* qrow = &qtile[nt & 1][rc * DCH];
    v8f s = {};
#pragma unroll
    for (int kk = 0; kk < 8; ++kk)
      s = wmma_f16(loadAfrag(qrow, kk, lane), bk[kk], s);
#pragma unroll
    for (int r = 0; r < 8; ++r) {        // fold 8 rows into this lane's column
      const float sv = s[r] + u[b * NPIX + n0 + r + rowoff];
      const float mn = fmaxf(mc, sv);
      rs = rs * __expf(mc - mn) + __expf(sv - mn);
      mc = mn;
    }
  }
  {                                      // lanes L and L+16 share a column
    const float m2 = __shfl_xor(mc, 16, 32);
    const float r2 = __shfl_xor(rs, 16, 32);
    const float mn = fmaxf(mc, m2);
    rs = rs * __expf(mc - mn) + r2 * __expf(m2 - mn);
    mc = mn;
  }
  if (lane < 16) {
    const float logm = logf(1.0f / (float)NPIX + 1e-8f);
    vp[b * NPIX + m0 + lane] = logm - (mc + logf(rs));
  }
}

// =====================================================================
// Kernel 4: out[n,:] = sum_m exp(negC + u + vpot) * v[m,:]  (pass 3).
// Block = 4 waves over the same 16 rows n; wave w owns d-slice [64w,64w+64).
// Shared 32-row K tile staged via TDM (double-buffered). Score tiles via
// WMMA; exp'd P bounced through per-wave LDS into A-layout; V fragments
// are contiguous vector loads from the d-major v tensor.
// =====================================================================
__global__ void sinkq_apply(const _Float16* __restrict__ qh,
                            const _Float16* __restrict__ kh,
                            const _Float16* __restrict__ vt,
                            const float* __restrict__ u,
                            const float* __restrict__ vp,
                            float* __restrict__ out) {
  __shared__ __align__(32) _Float16 ktile[2][32 * DCH];   // 2 x 16KB
  __shared__ __align__(32) _Float16 plds[4][16][40];      // per-wave P tile

  const int tid  = threadIdx.x;
  const int lane = tid & 31;
  const int wave = tid >> 5;            // 0..3 -> d-slice
  const int p0 = blockIdx.x * 16;
  const int b  = p0 >> 12;
  const int n0 = p0 & (NPIX - 1);
  const int d0 = wave * 64;
  const int rc = lane & 15;
  const int rowoff = (lane >= 16) ? 8 : 0;
  const size_t qb = (size_t)b * NPIX * DCH;
  const size_t vb = (size_t)b * DCH * NPIX;

  v16h aq[8];
  const _Float16* qrow = qh + qb + (size_t)(n0 + rc) * DCH;
#pragma unroll
  for (int kk = 0; kk < 8; ++kk) aq[kk] = loadAfrag(qrow, kk, lane);

  float urow[8];
#pragma unroll
  for (int r = 0; r < 8; ++r) urow[r] = u[b * NPIX + n0 + r + rowoff];

  v8f z = {};
  v8f oacc[4] = { z, z, z, z };

#if HAVE_TDM
  if (wave == 0) tdm_load_tile(kh + qb, &ktile[0][0], 32);
#else
#pragma unroll
  for (int i = 0; i < 4; ++i)
    *(v16h*)&ktile[0][(i * 128 + tid) * 16] = *(const v16h*)&kh[qb + (i * 128 + tid) * 16];
#endif

  for (int mt = 0; mt < NPIX / 32; ++mt) {
#if HAVE_TDM
    if (wave == 0) tdm_wait();
#endif
    __syncthreads();
    if (mt + 1 < NPIX / 32) {
#if HAVE_TDM
      if (wave == 0)
        tdm_load_tile(kh + qb + (size_t)(mt + 1) * 32 * DCH,
                      &ktile[(mt + 1) & 1][0], 32);
#else
      const _Float16* src = kh + qb + (size_t)(mt + 1) * 32 * DCH;
#pragma unroll
      for (int i = 0; i < 4; ++i)
        *(v16h*)&ktile[(mt + 1) & 1][(i * 128 + tid) * 16] =
            *(const v16h*)&src[(i * 128 + tid) * 16];
      __builtin_prefetch(&kh[qb + (size_t)(mt + 2) * 32 * DCH + tid * 64], 0, 0);
#endif
    }
    const int m0 = mt * 32;
    const _Float16* k0 = &ktile[mt & 1][rc * DCH];
    const _Float16* k1 = &ktile[mt & 1][(16 + rc) * DCH];

    v8f s0 = {}, s1 = {};
#pragma unroll
    for (int kk = 0; kk < 8; ++kk) {
      s0 = wmma_f16(aq[kk], loadBfrag(k0, kk, lane), s0);
      s1 = wmma_f16(aq[kk], loadBfrag(k1, kk, lane), s1);
    }

    const float vp0 = vp[b * NPIX + m0 + rc];
    const float vp1 = vp[b * NPIX + m0 + 16 + rc];
#pragma unroll
    for (int r = 0; r < 8; ++r) {                 // P = exp(S + u + vpot)
      plds[wave][r + rowoff][rc]      = (_Float16)__expf(s0[r] + urow[r] + vp0);
      plds[wave][r + rowoff][16 + rc] = (_Float16)__expf(s1[r] + urow[r] + vp1);
    }
    // per-wave LDS region; LDS ops in-order within a wave -> no barrier
    const v16h ap = loadAfrag(&plds[wave][rc][0], 0, lane);

#pragma unroll
    for (int t = 0; t < 4; ++t) {                 // P(16x32) x V(32x16)
      const v16h bf = *(const v16h*)&vt[vb + (size_t)(d0 + t * 16 + rc) * NPIX +
                                       m0 + ((lane >= 16) ? 16 : 0)];
      oacc[t] = wmma_f16(ap, bf, oacc[t]);
    }
  }

#pragma unroll
  for (int t = 0; t < 4; ++t)
#pragma unroll
    for (int r = 0; r < 8; ++r) {
      const int dd = d0 + t * 16 + rc;
      const int n  = n0 + r + rowoff;
      out[(size_t)b * DCH * NPIX + (size_t)dd * NPIX + n] = oacc[t][r];
    }
}

// =====================================================================
extern "C" void kernel_launch(void* const* d_in, const int* in_sizes, int n_in,
                              void* d_out, int out_size, void* d_ws, size_t ws_size,
                              hipStream_t stream) {
  const float* x    = (const float*)d_in[0];   // [B, D, H, W]
  const float* Wt   = (const float*)d_in[1];   // [3D, D]
  const float* bias = (const float*)d_in[2];   // [3D]
  float* out = (float*)d_out;                  // [B, D, H, W]

  const size_t nqkv = (size_t)BATCH * NPIX * DCH;
  _Float16* qh = (_Float16*)d_ws;              // [B][N][D]
  _Float16* kh = qh + nqkv;                    // [B][N][D]
  _Float16* vt = kh + nqkv;                    // [B][D][N] (transposed)
  float* u  = (float*)(vt + nqkv);             // [B][N]
  float* vp = u + BATCH * NPIX;                // [B][N]

  sinkq_qkv<<<dim3(OCH / 128, (BATCH * NPIX) / 16), 256, 0, stream>>>(
      x, Wt, bias, qh, kh, vt);
  sinkq_row_lse<<<dim3((BATCH * NPIX / 16) / 8), 256, 0, stream>>>(qh, kh, u);
  sinkq_col_lse<<<dim3((BATCH * NPIX / 16) / 8), 256, 0, stream>>>(qh, kh, u, vp);
  sinkq_apply<<<dim3(BATCH * NPIX / 16), 128, 0, stream>>>(qh, kh, vt, u, vp, out);
}
